// Conv2DMod_4174708211799
// MI455X (gfx1250) — matrix operand order repaired and checked by reading, not compiled, not run on hardware
//
#include <hip/hip_runtime.h>
#include <stdint.h>

typedef __bf16 bf16_t;
typedef __attribute__((ext_vector_type(16))) __bf16 v16bf;
typedef __attribute__((ext_vector_type(8)))  __bf16 v8bf;
typedef __attribute__((ext_vector_type(8)))  float  v8f;

#define B_    8
#define C_    256
#define F_    256
#define H_    128
#define W_    128
#define HP_   130            // H padded by one row top/bottom
#define SDIM  512
#define LDA_STRIDE 40        // bf16/row: 32 data + 8 pad, keeps 16B alignment
#define LDB_STRIDE 40

// ---------------- Kernel 1: latent = s @ affine_w^T + affine_b, store (latent+1) ----
__global__ void k_mod(const float* __restrict__ s, const float* __restrict__ aw,
                      const float* __restrict__ ab, float* __restrict__ mod) {
    const int b = blockIdx.x;
    const int c = threadIdx.x;
    const float* sp = s  + (size_t)b * SDIM;
    const float* wp = aw + (size_t)c * SDIM;
    float acc = ab[c];
    #pragma unroll 8
    for (int j = 0; j < SDIM; ++j) acc += sp[j] * wp[j];
    mod[b * C_ + c] = acc + 1.0f;
}

// ---------------- Kernel 2: modulate, demodulate, emit bf16 weights -----------------
// Output layout: wmod[b][kidx=kh*3+kw][f][c]  (c contiguous)
__global__ void k_demod(const float* __restrict__ weight, const float* __restrict__ mod,
                        bf16_t* __restrict__ wmod) {
    const int b = blockIdx.x >> 8;
    const int f = blockIdx.x & 255;
    const int c = threadIdx.x;

    const float m = mod[b * C_ + c];
    const float* wp = weight + ((size_t)f * C_ + c) * 9;

    float w9[9];
    float p = 0.f;
    #pragma unroll
    for (int i = 0; i < 9; ++i) { w9[i] = wp[i] * m; p += w9[i] * w9[i]; }

    __shared__ float red[256];
    red[c] = p;
    __syncthreads();
    for (int st = 128; st > 0; st >>= 1) {
        if (c < st) red[c] += red[c + st];
        __syncthreads();
    }
    const float d = rsqrtf(red[0] + 1e-8f);

    #pragma unroll
    for (int i = 0; i < 9; ++i)
        wmod[(((size_t)b * 9 + i) * F_ + f) * C_ + c] = (bf16_t)(w9[i] * d);
}

// ---------------- Kernel 3: x -> bf16 with one-row zero padding top/bottom ----------
// xbf[b][c][hp 0..129][w], hp=0 and hp=129 are zero rows (h border pad).
__global__ void k_cvt(const float* __restrict__ x, bf16_t* __restrict__ xbf) {
    const int w  = threadIdx.x;
    const int hp = blockIdx.x;
    const int c  = blockIdx.y;
    const int b  = blockIdx.z;
    float v = 0.f;
    if (hp > 0 && hp < HP_ - 1)
        v = x[(((size_t)b * C_ + c) * H_ + (hp - 1)) * W_ + w];
    xbf[(((size_t)b * C_ + c) * HP_ + hp) * W_ + w] = (bf16_t)v;
}

// ---------------- Kernel 4: implicit-GEMM conv via bf16 WMMA ------------------------
// grid = (H, 2, B); block = 256 (8 waves). Block tile: M=128 (f), N=128 (one w row).
// Per (kh, c0): stage 3 kw weight tiles (async DMA -> LDS) + one padded x tile,
// then 3 kw passes of 8 WMMAs per wave sharing the x tile via row-shifted reads.
__global__ __launch_bounds__(256)
void k_conv(const bf16_t* __restrict__ xbf, const bf16_t* __restrict__ wmod,
            float* __restrict__ out) {
    __shared__ bf16_t smemA[3 * 128 * LDA_STRIDE];  // [kw][f_local][c_local]
    __shared__ bf16_t smemB[HP_ * LDB_STRIDE];      // [w_pad 0..129][c_local]

    const int h  = blockIdx.x;
    const int f0 = blockIdx.y * 128;
    const int b  = blockIdx.z;

    const int tid  = threadIdx.x;
    const int lane = tid & 31;
    const int wave = tid >> 5;       // 0..7
    const int wm   = wave & 3;       // M-wave: rows wm*32..+31
    const int wn   = wave >> 2;      // N-wave: cols wn*64..+63
    const int hl   = lane & 15;
    const int hi   = lane >> 4;      // half-wave select

    v8f acc[2][4];
    #pragma unroll
    for (int i = 0; i < 2; ++i)
        #pragma unroll
        for (int j = 0; j < 4; ++j)
            acc[i][j] = (v8f){0.f, 0.f, 0.f, 0.f, 0.f, 0.f, 0.f, 0.f};

    // staging thread mappings
    const int fa_row = tid >> 1;          // 0..127 : A tile row (f)
    const int fa_chk = (tid & 1) * 16;    // 0|16   : A tile c-chunk (16 bf16 = 32B)
    const int cB     = tid >> 3;          // 0..31  : B tile channel
    const int wch    = (tid & 7) * 16;    // 0..112 : B tile w-chunk (16 w)

    // W-border pad rows (w_src = -1 and 128): zero once, constant for whole block
    if (tid < 32)      smemB[tid] = (bf16_t)0.f;
    else if (tid < 64) smemB[(HP_ - 1) * LDB_STRIDE + (tid - 32)] = (bf16_t)0.f;

    for (int kh = 0; kh < 3; ++kh) {
        const int hp = h + kh;  // padded source row, always valid in [0,129]
        for (int c0 = 0; c0 < C_; c0 += 32) {
            // ---- A: async-DMA three kw weight tiles (128x32 bf16 each) into LDS
            #pragma unroll
            for (int t3 = 0; t3 < 3; ++t3) {
                const bf16_t* g = wmod +
                    ((((size_t)b * 9 + kh * 3 + t3) * F_ + f0 + fa_row) * C_) + c0 + fa_chk;
                uint32_t lo = (uint32_t)(uintptr_t)
                    &smemA[(t3 * 128 + fa_row) * LDA_STRIDE + fa_chk];
                asm volatile("global_load_async_to_lds_b128 %0, %1, off"
                             :: "v"(lo), "v"((uint64_t)(uintptr_t)g) : "memory");
                asm volatile("global_load_async_to_lds_b128 %0, %1, off"
                             :: "v"(lo + 16u), "v"((uint64_t)(uintptr_t)(g + 8)) : "memory");
            }
            // ---- B: coalesced bf16 row loads, transposed scatter into [w][c] tile
            {
                const bf16_t* gx = xbf +
                    (((size_t)b * C_ + c0 + cB) * HP_ + hp) * W_ + wch;
                v8bf r0 = *(const v8bf*)(gx);
                v8bf r1 = *(const v8bf*)(gx + 8);
                #pragma unroll
                for (int e = 0; e < 8; ++e) {
                    smemB[(wch + e + 1) * LDB_STRIDE + cB] = r0[e];
                    smemB[(wch + e + 9) * LDB_STRIDE + cB] = r1[e];
                }
                if (c0 + 32 < C_)
                    __builtin_prefetch(gx + (size_t)32 * HP_ * W_, 0, 1);
            }
            asm volatile("s_wait_asynccnt 0x0" ::: "memory");
            __syncthreads();

            // ---- compute: 3 kw taps share the B tile via row-shifted reads
            const int akl = hi ? 16 : 0;   // A: lanes 0-15 K 0..15; lanes 16-31 K 16..31
            const int bkl = hi ? 16 : 0;   // B: lanes 0-15 K 0..15; lanes 16-31 K 16..31
            #pragma unroll
            for (int kwi = 0; kwi < 3; ++kwi) {
                v16bf afrag[2];
                #pragma unroll
                for (int i = 0; i < 2; ++i) {
                    const bf16_t* ap =
                        &smemA[(kwi * 128 + wm * 32 + i * 16 + hl) * LDA_STRIDE + akl];
                    v8bf l0 = *(const v8bf*)(ap);
                    v8bf l1 = *(const v8bf*)(ap + 8);
                    afrag[i] = __builtin_shufflevector(l0, l1,
                        0,1,2,3,4,5,6,7,8,9,10,11,12,13,14,15);
                }
                v16bf bfrag[4];
                #pragma unroll
                for (int j = 0; j < 4; ++j) {
                    const bf16_t* bp =
                        &smemB[(wn * 64 + j * 16 + hl + kwi) * LDB_STRIDE + bkl];
                    v8bf l0 = *(const v8bf*)(bp);
                    v8bf l1 = *(const v8bf*)(bp + 8);
                    bfrag[j] = __builtin_shufflevector(l0, l1,
                        0,1,2,3,4,5,6,7,8,9,10,11,12,13,14,15);
                }
                #pragma unroll
                for (int i = 0; i < 2; ++i)
                    #pragma unroll
                    for (int j = 0; j < 4; ++j)
                        acc[i][j] = __builtin_amdgcn_wmma_f32_16x16x32_bf16(
                            false, afrag[i], false, bfrag[j],
                            (short)0, acc[i][j], false, false);
            }
            __syncthreads();
        }
    }

    // ---- epilogue: C/D layout -> VGPR r holds rows M=r (lanes 0-15) and M=r+8 (16-31)
    float* obase = out + ((size_t)b * F_ * H_ + (size_t)h) * W_;
    #pragma unroll
    for (int i = 0; i < 2; ++i) {
        #pragma unroll
        for (int j = 0; j < 4; ++j) {
            const int wcol = wn * 64 + j * 16 + hl;
            #pragma unroll
            for (int r = 0; r < 8; ++r) {
                const int f = f0 + wm * 32 + i * 16 + r + hi * 8;
                obase[(size_t)f * H_ * W_ + wcol] = acc[i][j][r];
            }
        }
    }
}

// ---------------- launcher ----------------------------------------------------------
extern "C" void kernel_launch(void* const* d_in, const int* in_sizes, int n_in,
                              void* d_out, int out_size, void* d_ws, size_t ws_size,
                              hipStream_t stream) {
    const float* x        = (const float*)d_in[0];
    const float* s        = (const float*)d_in[1];
    const float* weight   = (const float*)d_in[2];
    const float* affine_w = (const float*)d_in[3];
    const float* affine_b = (const float*)d_in[4];
    float* out = (float*)d_out;

    float*  mod  = (float*)d_ws;                               // 8 KB
    bf16_t* wmod = (bf16_t*)((char*)d_ws + 8192);              // 9.4 MB
    bf16_t* xbf  = (bf16_t*)((char*)d_ws + 8192 + 9437184);    // 68.2 MB

    k_mod  <<<B_, C_, 0, stream>>>(s, affine_w, affine_b, mod);
    k_demod<<<B_ * F_, C_, 0, stream>>>(weight, mod, wmod);

    dim3 gcvt(HP_, C_, B_);
    k_cvt  <<<gcvt, W_, 0, stream>>>(x, xbf);

    dim3 grid(H_, F_ / 128, B_);
    k_conv <<<grid, 256, 0, stream>>>(xbf, wmod, out);
}
